// mymodel_58677843198444
// MI455X (gfx1250) — matrix operand order repaired and checked by reading, not compiled, not run on hardware
//
#include <hip/hip_runtime.h>

// ---------------------------------------------------------------------------
// Types for CDNA5 WMMA (wave32, v_wmma_f32_16x16x32_bf16)
// ---------------------------------------------------------------------------
typedef __bf16 bf16_t;
typedef bf16_t v16bf __attribute__((ext_vector_type(16)));
typedef bf16_t v8bf  __attribute__((ext_vector_type(8)));
typedef float  v8f   __attribute__((ext_vector_type(8)));
typedef unsigned int u32x4 __attribute__((ext_vector_type(4)));
typedef int i32x8 __attribute__((ext_vector_type(8)));
typedef int i32x4 __attribute__((ext_vector_type(4)));

union FragBF { v16bf v; v8bf h[2]; };

#define D_MODEL 256
#define H_HEADS 8
#define HD 32
#define EPS_LN 1e-5f
#define EPS_ATT 1e-6f
#define INV_LOG 0.03597789550970486f   // ln(10000)/256

// Epilogue mode bits
#define EP_BIAS 1
#define EP_PE   2
#define EP_RELU 4
#define EP_F32  8
#define EP_BF16 16

#if __has_builtin(__builtin_amdgcn_tensor_load_to_lds) && \
    __has_builtin(__builtin_amdgcn_s_wait_tensorcnt)
#define USE_TDM 1
#else
#define USE_TDM 0
#endif

// ---------------------------------------------------------------------------
// Generic tiled GEMM:  C[M,Ncol] = A[M,K](bf16,row-major) * Bt[Ncol,K](bf16)
// Block: 256 threads = 8 waves; wave w -> rows [m0 + 32w, +32) (2 A-frags);
// 4 N-tiles/wave (64 cols). B tile [64][K] staged in LDS via TDM
// (tensor_load_to_lds + s_wait_tensorcnt), rows padded +8 elems (16B) for
// bank-conflict-free fragment reads. Per k-step: 4 global + 8 ds b128 loads
// feed 8 back-to-back WMMAs (each B frag reused by 2 row-groups).
// ---------------------------------------------------------------------------
template<int MODE, int K>
__global__ __launch_bounds__(256) void gemm_wmma(
    const bf16_t* __restrict__ A, const bf16_t* __restrict__ Bt,
    const float* __restrict__ bias,
    float* __restrict__ Cf, bf16_t* __restrict__ Cb,
    int M, int Ncol, int reluLimit)
{
  constexpr int KP = K + 8;                      // padded LDS row stride
  extern __shared__ bf16_t smem[];               // [64][KP]

  const int lane = threadIdx.x & 31;
  const int wave = threadIdx.x >> 5;
  const int m0   = blockIdx.x * 256 + wave * 32;
  const int n0   = blockIdx.y * 64;
  const int l    = lane & 15;
  const int hi   = lane >> 4;                    // 0: lanes 0-15, 1: 16-31

  // ---- stage B tile [64][K] -> LDS -------------------------------------
#if USE_TDM
  if (wave == 0) {
    unsigned long long ga =
        (unsigned long long)(size_t)(Bt + (size_t)n0 * K);
    unsigned lds_base = (unsigned)(size_t)(void*)smem;  // LDS byte offset
    // pad: after K*2/4 dwords insert 4 dwords (16B = 8 elems)
    const int pad_interval = (K == 128) ? 5 : 6;        // 64 / 128 dwords
    const int pad_amount   = 3;                          // 4 dwords
    u32x4 g0;
    g0[0] = 1u;                                          // count=1
    g0[1] = lds_base;
    g0[2] = (unsigned)(ga & 0xFFFFFFFFu);
    g0[3] = (unsigned)((ga >> 32) & 0x01FFFFFFu) | (2u << 30);  // type=2
    i32x8 g1;
    g1[0] = (1 << 16) | (1 << 20) |                      // data_size=2B, pad_en
            (pad_interval << 22) | (pad_amount << 25);
    g1[1] = (int)((K & 0xFFFF) << 16);                   // tensor_dim0 lo
    g1[2] = (int)(((unsigned)K >> 16) | (64u << 16));    // dim0 hi | dim1 lo
    g1[3] = (int)((unsigned)K << 16);                    // dim1 hi=0 | tile0
    g1[4] = 64;                                          // tile1 | tile2=0
    g1[5] = K;                                           // dim0_stride lo
    g1[6] = 0;
    g1[7] = 0;
    i32x4 gz4 = {0, 0, 0, 0};
    i32x8 gz8 = {0, 0, 0, 0, 0, 0, 0, 0};
    __builtin_amdgcn_tensor_load_to_lds(g0, g1, gz4, gz4, gz8, 0);
    __builtin_amdgcn_s_wait_tensorcnt(0);
  }
  __syncthreads();
#else
  for (int idx = threadIdx.x; idx < 64 * (K / 8); idx += 256) {
    int r = idx / (K / 8), cc = idx % (K / 8);
    *(v8bf*)(smem + (size_t)r * KP + cc * 8) =
        *(const v8bf*)(Bt + (size_t)(n0 + r) * K + cc * 8);
  }
  __syncthreads();
#endif

  // ---- main K loop ------------------------------------------------------
  v8f acc[2][4] = {};

  int arow0 = m0 + l;      if (arow0 >= M) arow0 = M - 1;   // clamp, EXEC full
  int arow1 = m0 + 16 + l; if (arow1 >= M) arow1 = M - 1;
  const bf16_t* aptr0 = A + (size_t)arow0 * K + (hi << 3);
  const bf16_t* aptr1 = A + (size_t)arow1 * K + (hi << 3);
  const bf16_t* bptr  = smem + (size_t)l * KP + (hi << 3);

#pragma unroll
  for (int k0 = 0; k0 < K; k0 += 32) {
    FragBF a0, a1;
    a0.h[0] = *(const v8bf*)(aptr0 + k0);       // K = k0+{0..7}  (hi: 8..15)
    a0.h[1] = *(const v8bf*)(aptr0 + k0 + 16);  // K = k0+{16..23}(hi: 24..31)
    a1.h[0] = *(const v8bf*)(aptr1 + k0);
    a1.h[1] = *(const v8bf*)(aptr1 + k0 + 16);
    FragBF b[4];
#pragma unroll
    for (int t = 0; t < 4; ++t) {
      const bf16_t* bp = bptr + (size_t)(t * 16) * KP + k0;
      b[t].h[0] = *(const v8bf*)(bp);
      b[t].h[1] = *(const v8bf*)(bp + 16);
    }
#pragma unroll
    for (int t = 0; t < 4; ++t)
      acc[0][t] = __builtin_amdgcn_wmma_f32_16x16x32_bf16(
                    false, a0.v, false, b[t].v, (short)0, acc[0][t],
                    false, false);
#pragma unroll
    for (int t = 0; t < 4; ++t)
      acc[1][t] = __builtin_amdgcn_wmma_f32_16x16x32_bf16(
                    false, a1.v, false, b[t].v, (short)0, acc[1][t],
                    false, false);
  }

  // ---- epilogue ---------------------------------------------------------
#pragma unroll
  for (int g = 0; g < 2; ++g) {
    const int rbase = m0 + g * 16 + hi * 8;
#pragma unroll
    for (int t = 0; t < 4; ++t) {
      const int col = n0 + t * 16 + l;
      float bv = 0.0f;
      if (MODE & EP_BIAS) bv = bias[col];
#pragma unroll
      for (int r = 0; r < 8; ++r) {
        const int rowi = rbase + r;
        if (rowi < M) {
          float v = acc[g][t][r] + bv;
          if (MODE & EP_PE) {
            float ang = (float)rowi * __expf((float)(col & ~1) * (-INV_LOG));
            v += (col & 1) ? __cosf(ang) : __sinf(ang);
          }
          if (MODE & EP_RELU) { if (col < reluLimit) v = fmaxf(v, 0.0f); }
          if (MODE & EP_F32)  Cf[(size_t)rowi * Ncol + col] = v;
          if (MODE & EP_BF16) Cb[(size_t)rowi * Ncol + col] = (bf16_t)v;
        }
      }
    }
  }
}

// ---------------------------------------------------------------------------
// Small utility kernels
// ---------------------------------------------------------------------------
__global__ void transpose_to_bf16(const float* __restrict__ W,
                                  bf16_t* __restrict__ Wt, int K, int Ncol)
{
  int idx = blockIdx.x * blockDim.x + threadIdx.x;
  int total = K * Ncol;
  for (; idx < total; idx += gridDim.x * blockDim.x) {
    int k = idx / Ncol, n = idx % Ncol;
    Wt[(size_t)n * K + k] = (bf16_t)W[idx];
  }
}

__global__ void f32_to_bf16(const float* __restrict__ src,
                            bf16_t* __restrict__ dst, int n)
{
  int i = blockIdx.x * blockDim.x + threadIdx.x;
  for (; i < n; i += gridDim.x * blockDim.x) dst[i] = (bf16_t)src[i];
}

__global__ void relu_to_bf16(const float* __restrict__ src,
                             bf16_t* __restrict__ dst, int n)
{
  int i = blockIdx.x * blockDim.x + threadIdx.x;
  for (; i < n; i += gridDim.x * blockDim.x) dst[i] = (bf16_t)fmaxf(src[i], 0.0f);
}

__global__ void fill_f32(float* __restrict__ p, float v, int n)
{
  int i = blockIdx.x * blockDim.x + threadIdx.x;
  for (; i < n; i += gridDim.x * blockDim.x) p[i] = v;
}

// ---------------------------------------------------------------------------
// Linear attention: kv[h][d][e] = sum_n k[n,h,d]*v[n,h,e]; ksum[h][d] = sum k.
// ---------------------------------------------------------------------------
__global__ __launch_bounds__(256) void kv_reduce(
    const bf16_t* __restrict__ qkv, float* __restrict__ kv,
    float* __restrict__ ksum, int M)
{
  const int lane = threadIdx.x & 31;
  const int head = threadIdx.x >> 5;
  float acc[HD] = {};
  float accks = 0.0f;
  for (int n = blockIdx.x; n < M; n += gridDim.x) {
    const bf16_t* row = qkv + (size_t)n * 768;
    float k_l = (float)row[256 + head * HD + lane];
    float v_l = (float)row[512 + head * HD + lane];
    accks += k_l;
#pragma unroll
    for (int d = 0; d < HD; ++d)
      acc[d] += __shfl(k_l, d, 32) * v_l;
  }
#pragma unroll
  for (int d = 0; d < HD; ++d)
    atomicAdd(&kv[(head * HD + d) * HD + lane], acc[d]);
  atomicAdd(&ksum[head * HD + lane], accks);
}

__global__ __launch_bounds__(256) void att_combine(
    const bf16_t* __restrict__ qkv, const float* __restrict__ kv,
    const float* __restrict__ ksum, bf16_t* __restrict__ attv, int M)
{
  const int lane = threadIdx.x & 31;
  const int head = threadIdx.x >> 5;
  float kvcol[HD];
#pragma unroll
  for (int d = 0; d < HD; ++d)
    kvcol[d] = kv[(head * HD + d) * HD + lane];
  const float ks_l = ksum[head * HD + lane];

  for (int n = blockIdx.x; n < M; n += gridDim.x) {
    float q_l = (float)qkv[(size_t)n * 768 + head * HD + lane];
    float den = q_l * ks_l;
#pragma unroll
    for (int off = 16; off > 0; off >>= 1) den += __shfl_xor(den, off, 32);
    float num = 0.0f;
#pragma unroll
    for (int d = 0; d < HD; ++d)
      num += __shfl(q_l, d, 32) * kvcol[d];
    attv[(size_t)n * D_MODEL + head * HD + lane] = (bf16_t)(num / (den + EPS_ATT));
  }
}

// ---------------------------------------------------------------------------
// Wave-per-row LayerNorm + residual: out = res + LN(x)*g + b   (D = 256)
// ---------------------------------------------------------------------------
__global__ __launch_bounds__(256) void ln_residual(
    const float* __restrict__ x, const float* __restrict__ res,
    const float* __restrict__ g, const float* __restrict__ b,
    float* __restrict__ outf, bf16_t* __restrict__ outb, int M)
{
  const int row  = blockIdx.x * 8 + (threadIdx.x >> 5);
  if (row >= M) return;
  const int lane = threadIdx.x & 31;
  const float4* xr = (const float4*)(x + (size_t)row * D_MODEL);
  float4 v0 = xr[lane], v1 = xr[lane + 32];
  float s  = v0.x + v0.y + v0.z + v0.w + v1.x + v1.y + v1.z + v1.w;
  float ss = v0.x*v0.x + v0.y*v0.y + v0.z*v0.z + v0.w*v0.w
           + v1.x*v1.x + v1.y*v1.y + v1.z*v1.z + v1.w*v1.w;
#pragma unroll
  for (int off = 16; off > 0; off >>= 1) {
    s  += __shfl_xor(s,  off, 32);
    ss += __shfl_xor(ss, off, 32);
  }
  const float mu  = s  * (1.0f / D_MODEL);
  const float var = ss * (1.0f / D_MODEL) - mu * mu;
  const float inv = rsqrtf(var + EPS_LN);

  const float4* rr = (const float4*)(res + (size_t)row * D_MODEL);
  float* of = outf + (size_t)row * D_MODEL;
  bf16_t* ob = outb ? outb + (size_t)row * D_MODEL : nullptr;
  const float xv[8] = {v0.x,v0.y,v0.z,v0.w,v1.x,v1.y,v1.z,v1.w};
  float4 r0 = rr[lane], r1 = rr[lane + 32];
  const float rv[8] = {r0.x,r0.y,r0.z,r0.w,r1.x,r1.y,r1.z,r1.w};
#pragma unroll
  for (int j = 0; j < 8; ++j) {
    int c = (j < 4) ? lane * 4 + j : 128 + lane * 4 + (j - 4);
    float o = rv[j] + (xv[j] - mu) * inv * g[c] + b[c];
    of[c] = o;
    if (ob) ob[c] = (bf16_t)o;
  }
}

// ---------------------------------------------------------------------------
// Sparse normalized aggregation
// ---------------------------------------------------------------------------
__global__ void deg_accum(const int* __restrict__ row,
                          const float* __restrict__ w,
                          float* __restrict__ deg, int E)
{
  int e = blockIdx.x * blockDim.x + threadIdx.x;
  for (; e < E; e += gridDim.x * blockDim.x)
    atomicAdd(&deg[row[e]], w[e]);
}

__global__ void dinv_kernel(const float* __restrict__ deg,
                            float* __restrict__ dinv, int n)
{
  int i = blockIdx.x * blockDim.x + threadIdx.x;
  for (; i < n; i += gridDim.x * blockDim.x) {
    float d = deg[i];
    dinv[i] = (d > 0.0f) ? rsqrtf(d) : 0.0f;
  }
}

__global__ void agg_self(const float* __restrict__ src,
                         const float* __restrict__ dinv,
                         float* __restrict__ agg, int M)
{
  int idx = blockIdx.x * blockDim.x + threadIdx.x;
  int total = M * D_MODEL;
  for (; idx < total; idx += gridDim.x * blockDim.x) {
    int i = idx >> 8;
    float di = dinv[i];
    agg[idx] = di * di * src[idx];
  }
}

__global__ __launch_bounds__(256) void agg_edges(
    const int* __restrict__ erow, const int* __restrict__ ecol,
    const float* __restrict__ w, const float* __restrict__ dinv,
    const float* __restrict__ src, float* __restrict__ agg, int E)
{
  long long gid = (long long)blockIdx.x * blockDim.x + threadIdx.x;
  long long total = (long long)E * 64;
  for (; gid < total; gid += (long long)gridDim.x * blockDim.x) {
    int e = (int)(gid >> 6);
    int p = (int)(gid & 63);
    int r = erow[e], c = ecol[e];
    float coef = dinv[r] * w[e] * dinv[c];
    const float4 s = *(const float4*)(src + (size_t)c * D_MODEL + p * 4);
    float* dst = agg + (size_t)r * D_MODEL + p * 4;
    atomicAdd(dst + 0, coef * s.x);
    atomicAdd(dst + 1, coef * s.y);
    atomicAdd(dst + 2, coef * s.z);
    atomicAdd(dst + 3, coef * s.w);
  }
}

// ---------------------------------------------------------------------------
// Host-side orchestration
// ---------------------------------------------------------------------------
static inline char* carve(char*& p, size_t bytes) {
  char* r = p;
  p += (bytes + 255) & ~((size_t)255);
  return r;
}

extern "C" void kernel_launch(void* const* d_in, const int* in_sizes, int n_in,
                              void* d_out, int out_size, void* d_ws, size_t ws_size,
                              hipStream_t stream)
{
  const float* x     = (const float*)d_in[0];
  const int*   ei    = (const int*)  d_in[1];
  const float* ea    = (const float*)d_in[2];
  const float* W_in  = (const float*)d_in[4];
  const float* b_in  = (const float*)d_in[5];
  const float* Wqkv  = (const float*)d_in[6];
  const float* Wproj = (const float*)d_in[7];
  const float* g1    = (const float*)d_in[8];
  const float* be1   = (const float*)d_in[9];
  const float* W1    = (const float*)d_in[10];
  const float* bf1   = (const float*)d_in[11];
  const float* W2    = (const float*)d_in[12];
  const float* bf2   = (const float*)d_in[13];
  const float* g2    = (const float*)d_in[14];
  const float* be2   = (const float*)d_in[15];
  const float* W_out = (const float*)d_in[16];
  const float* b_out = (const float*)d_in[17];

  const int N = in_sizes[0] / 128;   // 50000
  const int E = in_sizes[2];         // 800000
  const int D_IN = 128, D = 256, DQKV = 768, D_OUT = 128;

  char* p = (char*)d_ws;
  bf16_t* Wt_in   = (bf16_t*)carve(p, (size_t)D * D_IN  * 2);
  bf16_t* Wt_qkv  = (bf16_t*)carve(p, (size_t)DQKV * D  * 2);
  bf16_t* Wt_proj = (bf16_t*)carve(p, (size_t)D * D     * 2);
  bf16_t* Wt_1    = (bf16_t*)carve(p, (size_t)D * D     * 2);
  bf16_t* Wt_2    = (bf16_t*)carve(p, (size_t)D * D     * 2);
  bf16_t* Wt_out  = (bf16_t*)carve(p, (size_t)D_OUT * D * 2);
  bf16_t* xb      = (bf16_t*)carve(p, (size_t)N * D_IN * 2);
  float*  h_f     = (float*) carve(p, (size_t)N * D * 4);
  bf16_t* h_b     = (bf16_t*)carve(p, (size_t)N * D * 2);
  bf16_t* qkv_b   = (bf16_t*)carve(p, (size_t)N * DQKV * 2);
  bf16_t* attv    = (bf16_t*)carve(p, (size_t)N * D * 2);
  float*  att_f   = (float*) carve(p, (size_t)N * D * 4);
  float*  src_f   = (float*) carve(p, (size_t)N * D * 4);
  float*  kv      = (float*) carve(p, (size_t)H_HEADS * HD * HD * 4);
  float*  ksum    = (float*) carve(p, (size_t)H_HEADS * HD * 4);
  float*  deg     = (float*) carve(p, (size_t)N * 4);
  float*  dinv    = (float*) carve(p, (size_t)N * 4);
  // Aliased (lifetimes disjoint):
  bf16_t* src_b   = h_b;               // reuse after qkv GEMM consumed h_b
  bf16_t* ff1_b   = attv;              // reuse after proj GEMM consumed attv
  float*  ff2_f   = att_f;             // reuse after LN1 consumed att_f
  float*  src2_f  = h_f;               // reuse after LN1 consumed h_f
  float*  agg_f   = (float*)qkv_b;     // reuse after att_combine consumed qkv
  bf16_t* aggb    = (bf16_t*)src_f;    // reuse after LN2 consumed src_f

  const int gxM = (N + 255) / 256;     // 196 M-tiles of 256
  dim3 blk(256);
  const size_t smem128 = 64 * (128 + 8) * sizeof(bf16_t);
  const size_t smem256 = 64 * (256 + 8) * sizeof(bf16_t);

  // 1) weight conversion / transpose
  transpose_to_bf16<<<64, blk, 0, stream>>>(W_in,  Wt_in,  D_IN, D);
  transpose_to_bf16<<<96, blk, 0, stream>>>(Wqkv,  Wt_qkv, D,    DQKV);
  transpose_to_bf16<<<64, blk, 0, stream>>>(Wproj, Wt_proj,D,    D);
  transpose_to_bf16<<<64, blk, 0, stream>>>(W1,    Wt_1,   D,    D);
  transpose_to_bf16<<<64, blk, 0, stream>>>(W2,    Wt_2,   D,    D);
  transpose_to_bf16<<<64, blk, 0, stream>>>(W_out, Wt_out, D,    D_OUT);
  f32_to_bf16<<<512, blk, 0, stream>>>(x, xb, N * D_IN);

  // 2) h = x @ W_in + b_in + PE      -> h_f (f32) and h_b (bf16)
  gemm_wmma<EP_BIAS|EP_PE|EP_F32|EP_BF16, 128>
      <<<dim3(gxM, D/64), blk, smem128, stream>>>(xb, Wt_in, b_in, h_f, h_b,
                                                  N, D, 0);

  // 3) qkv = h @ Wqkv, relu on q,k (cols < 512)  -> qkv_b (bf16)
  gemm_wmma<EP_RELU|EP_BF16, 256>
      <<<dim3(gxM, DQKV/64), blk, smem256, stream>>>(h_b, Wt_qkv, nullptr,
                                                     nullptr, qkv_b, N, DQKV, 512);

  // 4) linear attention reductions
  fill_f32<<<8, blk, 0, stream>>>(kv,   0.0f, H_HEADS * HD * HD);
  fill_f32<<<1, blk, 0, stream>>>(ksum, 0.0f, H_HEADS * HD);
  kv_reduce<<<256, blk, 0, stream>>>(qkv_b, kv, ksum, N);
  att_combine<<<512, blk, 0, stream>>>(qkv_b, kv, ksum, attv, N);

  // 5) att = attv @ Wproj  -> att_f
  gemm_wmma<EP_F32, 256>
      <<<dim3(gxM, D/64), blk, smem256, stream>>>(attv, Wt_proj, nullptr,
                                                  att_f, nullptr, N, D, 0);

  // 6) src = h + LN(att)   -> src_f + src_b
  ln_residual<<<(N + 7) / 8, blk, 0, stream>>>(att_f, h_f, g1, be1,
                                               src_f, src_b, N);

  // 7) ff1 = relu(src @ W1 + bf1) -> ff1_b ; ff2 = ff1 @ W2 + bf2 -> ff2_f
  gemm_wmma<EP_BIAS|EP_RELU|EP_BF16, 256>
      <<<dim3(gxM, D/64), blk, smem256, stream>>>(src_b, Wt_1, bf1,
                                                  nullptr, ff1_b, N, D, D);
  gemm_wmma<EP_BIAS|EP_F32, 256>
      <<<dim3(gxM, D/64), blk, smem256, stream>>>(ff1_b, Wt_2, bf2,
                                                  ff2_f, nullptr, N, D, 0);

  // 8) src2 = src + LN(ff2) -> src2_f
  ln_residual<<<(N + 7) / 8, blk, 0, stream>>>(ff2_f, src_f, g2, be2,
                                               src2_f, nullptr, N);

  // 9) symmetric normalization + sparse aggregation
  fill_f32<<<256, blk, 0, stream>>>(deg, 1.0f, N);          // self-loop weight 1
  deg_accum<<<1024, blk, 0, stream>>>(ei, ea, deg, E);       // + edge weights
  dinv_kernel<<<256, blk, 0, stream>>>(deg, dinv, N);
  agg_self<<<2048, blk, 0, stream>>>(src2_f, dinv, agg_f, N);
  {
    long long tot = (long long)E * 64;
    int blocks = (int)((tot + 255) / 256);
    agg_edges<<<blocks, blk, 0, stream>>>(ei, ei + E, ea, dinv,
                                          src2_f, agg_f, E);
  }

  // 10) out = relu(agg) @ W_out + b_out -> d_out (f32)
  relu_to_bf16<<<2048, blk, 0, stream>>>(agg_f, aggb, N * D);
  gemm_wmma<EP_BIAS|EP_F32, 256>
      <<<dim3(gxM, D_OUT/64), blk, smem256, stream>>>(aggb, Wt_out, b_out,
                                                      (float*)d_out, nullptr,
                                                      N, D_OUT, 0);
  (void)n_in; (void)out_size; (void)ws_size;
}
